// FlexAttentionWrapper_8564164788604
// MI455X (gfx1250) — matrix operand order repaired and checked by reading, not compiled, not run on hardware
//
#include <hip/hip_runtime.h>

typedef __attribute__((ext_vector_type(16))) __bf16 v16bf;
typedef __attribute__((ext_vector_type(8)))  __bf16 v8bf;
typedef __attribute__((ext_vector_type(8)))  float  v8f;
typedef __attribute__((ext_vector_type(4)))  float  f4;
typedef __attribute__((ext_vector_type(4)))  unsigned int u4;

#define H_HEADS 16
#define SEQ     8192
#define HDIM    64
#define WAVES   8          // 8 waves = 256 threads; block covers 128 queries
#define QBLK    (WAVES * 16)

#define KSTRIDE 72         // kbuf row stride in bf16 (64 used + pad; 144B, 16B-aligned rows)
#define VSTRIDE 40         // vbuf row stride in bf16 (32 used + pad; 80B, 16B-aligned rows)

__device__ __forceinline__ float row_max16(float x) {
  x = fmaxf(x, __shfl_xor(x, 1, 32));
  x = fmaxf(x, __shfl_xor(x, 2, 32));
  x = fmaxf(x, __shfl_xor(x, 4, 32));
  x = fmaxf(x, __shfl_xor(x, 8, 32));
  return x;
}

// load 16 bf16 (32B) from LDS into a WMMA operand vector (16B-aligned source)
__device__ __forceinline__ v16bf load_bf16x16(const __bf16* p) {
  v16bf r;
  *(u4*)&r       = *(const u4*)p;
  *((u4*)&r + 1) = *(const u4*)(p + 8);
  return r;
}

__global__ __launch_bounds__(32 * WAVES) void swa_fa_kernel(
    const float* __restrict__ Q, const float* __restrict__ K,
    const float* __restrict__ V, const int* __restrict__ wptr,
    float* __restrict__ O)
{
  const int w    = wptr[0];                      // 256 (device scalar; uniform)
  const int tid  = threadIdx.x;
  const int lane = tid & 31;
  const int wave = tid >> 5;
  const int h      = blockIdx.x >> 6;            // 64 blocks per head
  const int qblock = (blockIdx.x & 63) * QBLK;   // first query of block
  const int qbase  = qblock + wave * 16;         // this wave's 16 query rows
  const int half   = lane >> 4;
  const int qr     = lane & 15;

  // ---- LDS: block-shared bf16 K/V stage + per-wave P-transpose bounce ----
  __shared__ __align__(16) __bf16 kbuf[32 * KSTRIDE];     // [key][dim]
  __shared__ __align__(16) __bf16 vbuf[HDIM * VSTRIDE];   // [dim][key] (transposed)
  __shared__ __align__(16) float  pbuf_all[WAVES][16 * 36];
  float* pbuf = pbuf_all[wave];

  const float* Qh = Q + (size_t)h * SEQ * HDIM;
  const float* Kh = K + (size_t)h * SEQ * HDIM;
  const float* Vh = V + (size_t)h * SEQ * HDIM;

  // ---- Q tile into WMMA A layout, hi/lo bf16 split (once) ----
  // Fold D^-0.5 AND log2(e) into the Q scale: scores are then in the log2 domain,
  // so softmax uses the native v_exp_f32 (exp2) with no per-element multiplies.
  v16bf qhi[2], qlo[2];
  {
    const float* qrow = Qh + (size_t)(qbase + qr) * HDIM;
    const float scale = 0.125f * 1.44269504088896340736f;   // 64^-0.5 * log2(e)
#pragma unroll
    for (int c = 0; c < 2; ++c) {
#pragma unroll
      for (int e = 0; e < 16; ++e) {
        int d = c * 32 + half * 16 + e;
        float f = qrow[d] * scale;
        __bf16 hi = (__bf16)f;
        qhi[c][e] = hi;
        qlo[c][e] = (__bf16)(f - (float)hi);
      }
    }
  }

  // all-ones B operand: (P x ones) replicates row-sums into every C/D column
  v16bf onesb;
#pragma unroll
  for (int e = 0; e < 16; ++e) onesb[e] = (__bf16)1.0f;

  v8f acc0 = {}, acc1 = {}, acc2 = {}, acc3 = {};
  float m_r[8], l_r[8];
#pragma unroll
  for (int v = 0; v < 8; ++v) { m_r[v] = -3.0e38f; l_r[v] = 0.0f; }

  // block-uniform chunk range (union of the 8 waves' windows)
  int j_lo = qblock - w;           if (j_lo < 0)   j_lo = 0;
  int j_hi = qblock + QBLK + w;    if (j_hi > SEQ) j_hi = SEQ;

  // producer indexing: thread covers key pk (0..31), dims pd..pd+7
  const int pk = tid >> 3;
  const int pd = (tid & 7) * 8;

  for (int jg = j_lo; jg < j_hi; jg += 32) {
    // ================= producer: stage K,V chunk to LDS as bf16 =================
    {
      int kc = min(jg + pk, SEQ - 1);
      const f4* ks = (const f4*)(Kh + (size_t)kc * HDIM + pd);
      f4 ka = ks[0], kb4 = ks[1];
      v8bf kv;
#pragma unroll
      for (int i = 0; i < 4; ++i) {
        kv[i]     = (__bf16)ka[i];
        kv[4 + i] = (__bf16)kb4[i];
      }
      *(v8bf*)(kbuf + pk * KSTRIDE + pd) = kv;

      const f4* vs = (const f4*)(Vh + (size_t)kc * HDIM + pd);
      f4 va = vs[0], vb4 = vs[1];
#pragma unroll
      for (int i = 0; i < 4; ++i) {
        vbuf[(pd + i) * VSTRIDE + pk]     = (__bf16)va[i];
        vbuf[(pd + 4 + i) * VSTRIDE + pk] = (__bf16)vb4[i];
      }
    }
    __syncthreads();

    // ================= consumer: this wave's flash-attention step =================
    bool active = (jg + 31 >= qbase - w) && (jg <= qbase + 15 + w);
    if (active) {
      // ---- scores: S = (Q*scale) K^T (log2 domain), two 16-key tiles ----
      v8f s0 = {}, s1 = {};
#pragma unroll
      for (int t = 0; t < 2; ++t) {
        const __bf16* krow = kbuf + (t * 16 + qr) * KSTRIDE + half * 16;
#pragma unroll
        for (int c = 0; c < 2; ++c) {
          v16bf kb = load_bf16x16(krow + c * 32);
          if (t == 0) {
            s0 = __builtin_amdgcn_wmma_f32_16x16x32_bf16(false, qhi[c], false, kb, (short)0, s0, false, false);
            s0 = __builtin_amdgcn_wmma_f32_16x16x32_bf16(false, qlo[c], false, kb, (short)0, s0, false, false);
          } else {
            s1 = __builtin_amdgcn_wmma_f32_16x16x32_bf16(false, qhi[c], false, kb, (short)0, s1, false, false);
            s1 = __builtin_amdgcn_wmma_f32_16x16x32_bf16(false, qlo[c], false, kb, (short)0, s1, false, false);
          }
        }
      }

      // ---- band / range mask (edge chunks only; wave-uniform branch) ----
      bool interior = (jg >= qbase + 15 - w) && (jg + 31 <= qbase + w) && (jg + 31 < SEQ);
      if (!interior) {
#pragma unroll
        for (int v = 0; v < 8; ++v) {
          int q  = qbase + v + half * 8;
          int k0 = jg + qr;
          int k1 = jg + 16 + qr;
          bool ok0 = ((unsigned)(k0 - q + w) <= (unsigned)(2 * w)) && (k0 < SEQ);
          bool ok1 = ((unsigned)(k1 - q + w) <= (unsigned)(2 * w)) && (k1 < SEQ);
          if (!ok0) s0[v] = -1.0e30f;
          if (!ok1) s1[v] = -1.0e30f;
        }
      }

      // ---- online softmax (row = v + 8*half); m clamped so fully-masked rows give p=0 ----
      float corr[8], p0[8], p1[8];
#pragma unroll
      for (int v = 0; v < 8; ++v) {
        float mt = row_max16(fmaxf(s0[v], s1[v]));
        float mn = fmaxf(fmaxf(m_r[v], mt), -1.0e29f);
        corr[v] = __builtin_amdgcn_exp2f(m_r[v] - mn);
        m_r[v]  = mn;
        p0[v] = __builtin_amdgcn_exp2f(s0[v] - mn);
        p1[v] = __builtin_amdgcn_exp2f(s1[v] - mn);
        acc0[v] *= corr[v]; acc1[v] *= corr[v]; acc2[v] *= corr[v]; acc3[v] *= corr[v];
      }

      // ---- transpose P (C/D -> A layout) via wave-private LDS, hi/lo split ----
#pragma unroll
      for (int v = 0; v < 8; ++v) {
        int row = v + half * 8;
        pbuf[row * 36 + qr]      = p0[v];
        pbuf[row * 36 + 16 + qr] = p1[v];
      }
      asm volatile("" ::: "memory");   // per-wave LDS ops are in-order
      v16bf pAhi, pAlo;
      {
        const float* pr = pbuf + qr * 36 + half * 16;
        f4 a0 = ((const f4*)pr)[0];
        f4 a1 = ((const f4*)pr)[1];
        f4 a2 = ((const f4*)(pr + 8))[0];
        f4 a3 = ((const f4*)(pr + 8))[1];
#pragma unroll
        for (int e = 0; e < 4; ++e) {
          float f; __bf16 hi;
          f = a0[e]; hi = (__bf16)f; pAhi[e]      = hi; pAlo[e]      = (__bf16)(f - (float)hi);
          f = a1[e]; hi = (__bf16)f; pAhi[4 + e]  = hi; pAlo[4 + e]  = (__bf16)(f - (float)hi);
          f = a2[e]; hi = (__bf16)f; pAhi[8 + e]  = hi; pAlo[8 + e]  = (__bf16)(f - (float)hi);
          f = a3[e]; hi = (__bf16)f; pAhi[12 + e] = hi; pAlo[12 + e] = (__bf16)(f - (float)hi);
        }
      }

      // ---- row sums via WMMA against all-ones B (replicated into every column) ----
      v8f ls = {};
      ls = __builtin_amdgcn_wmma_f32_16x16x32_bf16(false, pAhi, false, onesb, (short)0, ls, false, false);
      ls = __builtin_amdgcn_wmma_f32_16x16x32_bf16(false, pAlo, false, onesb, (short)0, ls, false, false);
#pragma unroll
      for (int v = 0; v < 8; ++v) l_r[v] = l_r[v] * corr[v] + ls[v];

      // ---- O += P V, B operands straight from transposed LDS stage ----
#pragma unroll
      for (int nt = 0; nt < 4; ++nt) {
        v16bf vb = load_bf16x16(vbuf + (nt * 16 + qr) * VSTRIDE + half * 16);
        if (nt == 0) {
          acc0 = __builtin_amdgcn_wmma_f32_16x16x32_bf16(false, pAhi, false, vb, (short)0, acc0, false, false);
          acc0 = __builtin_amdgcn_wmma_f32_16x16x32_bf16(false, pAlo, false, vb, (short)0, acc0, false, false);
        } else if (nt == 1) {
          acc1 = __builtin_amdgcn_wmma_f32_16x16x32_bf16(false, pAhi, false, vb, (short)0, acc1, false, false);
          acc1 = __builtin_amdgcn_wmma_f32_16x16x32_bf16(false, pAlo, false, vb, (short)0, acc1, false, false);
        } else if (nt == 2) {
          acc2 = __builtin_amdgcn_wmma_f32_16x16x32_bf16(false, pAhi, false, vb, (short)0, acc2, false, false);
          acc2 = __builtin_amdgcn_wmma_f32_16x16x32_bf16(false, pAlo, false, vb, (short)0, acc2, false, false);
        } else {
          acc3 = __builtin_amdgcn_wmma_f32_16x16x32_bf16(false, pAhi, false, vb, (short)0, acc3, false, false);
          acc3 = __builtin_amdgcn_wmma_f32_16x16x32_bf16(false, pAlo, false, vb, (short)0, acc3, false, false);
        }
      }
    }
    __syncthreads();   // protect the LDS stage before next producer pass
  }

  // ---- normalize and store (C/D layout: row = v + 8*half, col dim = nt*16 + qr) ----
  float* obase = O + (size_t)(h * SEQ + qbase) * HDIM;
#pragma unroll
  for (int v = 0; v < 8; ++v) {
    float rl = 1.0f / l_r[v];
    float* orow = obase + (size_t)(v + half * 8) * HDIM;
    orow[0 * 16 + qr] = acc0[v] * rl;
    orow[1 * 16 + qr] = acc1[v] * rl;
    orow[2 * 16 + qr] = acc2[v] * rl;
    orow[3 * 16 + qr] = acc3[v] * rl;
  }
}

extern "C" void kernel_launch(void* const* d_in, const int* in_sizes, int n_in,
                              void* d_out, int out_size, void* d_ws, size_t ws_size,
                              hipStream_t stream) {
  const float* Q  = (const float*)d_in[0];
  const float* K  = (const float*)d_in[1];
  const float* V  = (const float*)d_in[2];
  const int*   wp = (const int*)d_in[4];     // window_size scalar on device
  float*       O  = (float*)d_out;

  dim3 grid(H_HEADS * (SEQ / QBLK));         // 16 * 64 = 1024 blocks
  dim3 block(32 * WAVES);                    // 256 threads = 8 waves
  swa_fa_kernel<<<grid, block, 0, stream>>>(Q, K, V, wp, O);
}
